// GRNet_12627203850881
// MI455X (gfx1250) — matrix operand order repaired and checked by reading, not compiled, not run on hardware
//
#include <hip/hip_runtime.h>
#include <hip/hip_bf16.h>

typedef __attribute__((ext_vector_type(16))) _Float16     v16h;
typedef __attribute__((ext_vector_type(8)))  float        v8f;
typedef __attribute__((ext_vector_type(2)))  float        v2f;
typedef __attribute__((ext_vector_type(4)))  unsigned int u32x4;
typedef __attribute__((ext_vector_type(8)))  int          i32x8;
typedef __attribute__((ext_vector_type(4)))  int          i32x4;

#if defined(__HIP_DEVICE_COMPILE__) && __has_builtin(__builtin_amdgcn_tensor_load_to_lds)
#define HAVE_TDM 1
#else
#define HAVE_TDM 0
#endif

// ---------------------------------------------------------------------------
// Utility kernels
// ---------------------------------------------------------------------------
__global__ void k_fill_zero(float* p, long n) {
  long i = (long)blockIdx.x * blockDim.x + threadIdx.x;
  if (i < n) p[i] = 0.f;
}

// ---------------------------------------------------------------------------
// Gridding: trilinear scatter of [-1,1]^3 points onto (128,128,16) grid
// ---------------------------------------------------------------------------
__global__ void k_gridding(const float* __restrict__ pc, float* __restrict__ grid) {
  int idx = blockIdx.x * blockDim.x + threadIdx.x;           // B*N = 4*2048
  if (idx >= 4 * 2048) return;
  int b = idx >> 11;
  const float* p = pc + (long)idx * 3;
  float gx = fminf(fmaxf((p[0] * 0.5f + 0.5f) * 128.f, 0.f), 126.999999f);
  float gy = fminf(fmaxf((p[1] * 0.5f + 0.5f) * 128.f, 0.f), 126.999999f);
  float gz = fminf(fmaxf((p[2] * 0.5f + 0.5f) * 16.f,  0.f), 14.999999f);
  float fx = floorf(gx), fy = floorf(gy), fz = floorf(gz);
  float dx = gx - fx, dy = gy - fy, dz = gz - fz;
  int ix = (int)fx, iy = (int)fy, iz = (int)fz;
  for (int c = 0; c < 8; ++c) {
    int i = c >> 2, j = (c >> 1) & 1, k = c & 1;
    float w = (i ? dx : 1.f - dx) * (j ? dy : 1.f - dy) * (k ? dz : 1.f - dz);
    atomicAdd(&grid[(((long)b * 128 + ix + i) * 128 + iy + j) * 16 + iz + k], w);
  }
}

// ---------------------------------------------------------------------------
// TDM: 2D tile load (Global -> LDS).  dim0 = contiguous row direction.
// tensor_dim* are the REMAINING elements from the tile origin, so hardware
// OOB handling zero-fills M/N/K tile edges for us (ISA 8.2).
// ---------------------------------------------------------------------------
#if HAVE_TDM
__device__ __forceinline__ void tdm_load_2d(const float* gaddr, unsigned lds_off_bytes,
                                            long dim0_rem, long dim1_rem,
                                            long stride_elems, int tile0, int tile1) {
  unsigned long long ga = (unsigned long long)(uintptr_t)gaddr;
  u32x4 g0;
  g0.x = 1u;                                            // count=1, user descriptor
  g0.y = lds_off_bytes;                                 // lds_addr [63:32]
  g0.z = (unsigned)(ga & 0xFFFFFFFFu);                  // global_addr lo
  g0.w = (unsigned)((ga >> 32) & 0x1FFFFFFu) | (2u << 30);  // addr hi | type=2
  unsigned td0 = (unsigned)(dim0_rem < 0 ? 0 : dim0_rem);
  unsigned td1 = (unsigned)(dim1_rem < 0 ? 0 : dim1_rem);
  unsigned long long st = (unsigned long long)stride_elems;
  i32x8 g1;
  g1[0] = (int)(2u << 16);                              // data_size = 4 bytes
  g1[1] = (int)((td0 & 0xFFFFu) << 16);                 // tensor_dim0[15:0]
  g1[2] = (int)(((td0 >> 16) & 0xFFFFu) | ((td1 & 0xFFFFu) << 16));
  g1[3] = (int)(((td1 >> 16) & 0xFFFFu) | ((unsigned)tile0 << 16));  // tile_dim0
  g1[4] = (int)((unsigned)tile1 & 0xFFFFu);             // tile_dim1 (tile_dim2=0)
  g1[5] = (int)(st & 0xFFFFFFFFu);                      // tensor_dim0_stride lo
  g1[6] = (int)((st >> 32) & 0xFFFFu);                  // stride hi (dim1_stride=0)
  g1[7] = 0;
  i32x4 z4 = {0, 0, 0, 0};
#if defined(__clang_major__) && __clang_major__ >= 23
  i32x8 z8 = {0, 0, 0, 0, 0, 0, 0, 0};
  __builtin_amdgcn_tensor_load_to_lds(g0, g1, z4, z4, z8, 0);
#else
  __builtin_amdgcn_tensor_load_to_lds(g0, g1, z4, z4, 0);
#endif
}
#endif

// ---------------------------------------------------------------------------
// Generic GEMM  C[M,N] = act(A[M,K] @ B[K,N] + bias) (+ res)
// V_WMMA_F32_16X16X32_F16, one wave per 16x16 tile, 4 waves/block.
// TDM double-buffers the 16x32 A panel and 32x16 B panel per wave in LDS:
//   tensor_load_to_lds -> s_wait_tensorcnt -> ds reads -> wmma
// ---------------------------------------------------------------------------
__global__ void __launch_bounds__(128)
k_gemm_f16_wmma(const float* __restrict__ A, const float* __restrict__ B,
                const float* __restrict__ bias, const float* __restrict__ res,
                float* __restrict__ C, int M, int N, int K, int act) {
  __shared__ float smem[8192];                 // 4 waves * 2 stages * 1024 floats
  int wid = threadIdx.x >> 5;
  int wave = (int)((blockIdx.x * (long)blockDim.x + threadIdx.x) >> 5);
  int lane = threadIdx.x & 31;
  int ntiles = (N + 15) >> 4;
  int mtiles = (M + 15) >> 4;
  if (wave >= ntiles * mtiles) return;         // wave-uniform: EXEC stays full
  int tm = wave / ntiles, tn = wave % ntiles;
  int m0 = tm << 4, n0 = tn << 4;
  int half_ = lane >> 4;
  int l = lane & 15;
  float* base = smem + wid * 2048;             // per-wave region (8 KB)
  v8f acc = {};

#if HAVE_TDM
  // LDS aperture: low 32 bits of the generic pointer are the LDS byte offset.
  unsigned lds_base = (unsigned)(uintptr_t)(void*)base;
  int iters = (K + 31) / 32;
  // stage s layout: A panel (16 rows x 32 k) at +s*4096B, B panel (32 k x 16) at +2048B
  auto issue = [&](int k0, int s) {
    tdm_load_2d(A + (long)m0 * K + k0, lds_base + (unsigned)s * 4096u,
                (long)K - k0, (long)M - m0, K, 32, 16);
    tdm_load_2d(B + (long)k0 * N + n0, lds_base + (unsigned)s * 4096u + 2048u,
                (long)N - n0, (long)K - k0, N, 16, 32);
  };
  issue(0, 0);
  for (int i = 0; i < iters; ++i) {
    if (i + 1 < iters) {                       // double-buffer: overlap next panel
      issue((i + 1) * 32, (i + 1) & 1);
      __builtin_amdgcn_s_wait_tensorcnt(2);    // wait for stage i's two loads
    } else {
      __builtin_amdgcn_s_wait_tensorcnt(0);
    }
    const float* As = base + (i & 1) * 1024;
    const float* Bs = As + 512;
    v16h a, b;
#pragma unroll
    for (int j = 0; j < 16; ++j) {             // A 16x32, ISA 7.12.2 layout
      int kl = (j < 8) ? (half_ * 8 + j) : (16 + half_ * 8 + (j - 8));
      a[j] = (_Float16)As[l * 32 + kl];
    }
#pragma unroll
    for (int j = 0; j < 16; ++j) {             // B 32x16, same K order as A
      int kl = (j < 8) ? (half_ * 8 + j) : (16 + half_ * 8 + (j - 8));
      b[j] = (_Float16)Bs[kl * 16 + l];
    }
    acc = __builtin_amdgcn_wmma_f32_16x16x32_f16(false, a, false, b, (short)0,
                                                 acc, false, false);
  }
#else
  // Fallback: direct global-gather fragments
  int am = m0 + l, bncol = n0 + l;
  for (int k0 = 0; k0 < K; k0 += 32) {
    v16h a, b;
#pragma unroll
    for (int j = 0; j < 16; ++j) {
      int kk = k0 + ((j < 8) ? (half_ * 8 + j) : (16 + half_ * 8 + (j - 8)));
      a[j] = (_Float16)((am < M && kk < K) ? A[(long)am * K + kk] : 0.f);
    }
#pragma unroll
    for (int j = 0; j < 16; ++j) {
      int kk = k0 + ((j < 8) ? (half_ * 8 + j) : (16 + half_ * 8 + (j - 8)));
      b[j] = (_Float16)((bncol < N && kk < K) ? B[(long)kk * N + bncol] : 0.f);
    }
    acc = __builtin_amdgcn_wmma_f32_16x16x32_f16(false, a, false, b, (short)0,
                                                 acc, false, false);
  }
#endif

  int bn = n0 + l;
  if (bn < N) {
#pragma unroll
    for (int r = 0; r < 8; ++r) {
      int m = m0 + half_ * 8 + r;
      if (m < M) {
        float v = acc[r] + (bias ? bias[bn] : 0.f);
        if (act == 2) v = fmaxf(v, 0.f);
        if (res) v += res[(long)m * N + bn];
        C[(long)m * N + bn] = v;
      }
    }
  }
}

// ---------------------------------------------------------------------------
// Implicit-GEMM 3x3x3 conv (pad 1, stride 1) on V_WMMA_F32_16X16X4_F32.
// Incremental base-3 tap decode (no div/mod in the K loop).
// ---------------------------------------------------------------------------
__global__ void __launch_bounds__(256)
k_conv3d_wmma(const float* __restrict__ x, const float* __restrict__ w,
              const float* __restrict__ bias, float* __restrict__ y,
              int NB, int Cin, int Cout, int D, int H, int W) {
  int wave = (int)((blockIdx.x * (long)blockDim.x + threadIdx.x) >> 5);
  int lane = threadIdx.x & 31;
  long NN = (long)NB * D * H * W;
  int ntiles = (int)((NN + 15) >> 4);
  int mtiles = (Cout + 15) >> 4;
  if (wave >= ntiles * mtiles) return;   // wave-uniform exit
  int tm = wave / ntiles, tn = wave % ntiles;
  int m0 = tm << 4, n0 = tn << 4;
  int half_ = lane >> 4;
  int l = lane & 15;

  long col = (long)n0 + l;
  bool colok = col < NN;
  int ow = 0, oh = 0, od = 0, nb = 0;
  if (colok) {
    long t = col;
    ow = (int)(t % W); t /= W;
    oh = (int)(t % H); t /= H;
    od = (int)(t % D); nb = (int)(t / D);
  }
  int K = Cin * 27;
  int am = m0 + l;
  long xbase = (long)nb * Cin * D * H * W;

  // per-lane tap state for the two K slots (advance by +4 per step)
  int ci0[2], kd0[2], kh0[2], kw0[2];
#pragma unroll
  for (int j = 0; j < 2; ++j) {
    int kk = half_ * 2 + j;
    ci0[j] = kk / 27;
    int r = kk % 27;
    kd0[j] = r / 9; kh0[j] = (r / 3) % 3; kw0[j] = r % 3;
  }

  v8f acc = {};
  for (int k0 = 0; k0 < K; k0 += 4) {
    v2f a, b;
#pragma unroll
    for (int j = 0; j < 2; ++j) {
      int kk = k0 + half_ * 2 + j;
      a[j] = (am < Cout && kk < K) ? w[(long)am * K + kk] : 0.f;
      float bv = 0.f;
      if (colok && kk < K) {
        int id = od + kd0[j] - 1, ih = oh + kh0[j] - 1, iw = ow + kw0[j] - 1;
        if ((unsigned)id < (unsigned)D && (unsigned)ih < (unsigned)H &&
            (unsigned)iw < (unsigned)W)
          bv = x[xbase + (((long)ci0[j] * D + id) * H + ih) * (long)W + iw];
      }
      b[j] = bv;
      // advance tap by 4 = +1 kh, +1 kw with base-3 carries
      kw0[j] += 1; if (kw0[j] >= 3) { kw0[j] -= 3; kh0[j] += 1; }
      kh0[j] += 1; if (kh0[j] >= 3) { kh0[j] -= 3; kd0[j] += 1; }
      if (kd0[j] >= 3) { kd0[j] -= 3; ci0[j] += 1; }
    }
    acc = __builtin_amdgcn_wmma_f32_16x16x4_f32(false, a, false, b, (short)0,
                                                acc, false, false);
  }
  if (colok) {
#pragma unroll
    for (int r = 0; r < 8; ++r) {
      int co = m0 + half_ * 8 + r;
      if (co < Cout)
        y[(((long)nb * Cout + co) * D + od) * (long)H * W + (long)oh * W + ow] =
            acc[r] + bias[co];
    }
  }
}

// ---------------------------------------------------------------------------
// BatchNorm (training-mode batch stats) + activation + optional residual
// ---------------------------------------------------------------------------
__global__ void k_bn_stats(const float* __restrict__ x, float* mean, float* var,
                           int C, long S, int NB) {
  int c = blockIdx.x;
  long total = (long)NB * S;
  float s = 0.f, ss = 0.f;
  for (long i = threadIdx.x; i < total; i += blockDim.x) {
    long nb = i / S, r = i - nb * S;
    float v = x[((long)nb * C + c) * S + r];
    s += v; ss += v * v;
  }
  __shared__ float sh1[256], sh2[256];
  sh1[threadIdx.x] = s; sh2[threadIdx.x] = ss;
  __syncthreads();
  for (int st = 128; st > 0; st >>= 1) {
    if ((int)threadIdx.x < st) {
      sh1[threadIdx.x] += sh1[threadIdx.x + st];
      sh2[threadIdx.x] += sh2[threadIdx.x + st];
    }
    __syncthreads();
  }
  if (threadIdx.x == 0) {
    float m = sh1[0] / (float)total;
    mean[c] = m;
    var[c] = sh2[0] / (float)total - m * m;
  }
}

// act: 0 none, 1 lrelu(0.2), 2 relu
__global__ void k_bn_act_res(const float* __restrict__ x, float* __restrict__ y,
                             const float* mean, const float* var,
                             const float* g, const float* beta, int C, long S,
                             long total, int act, const float* __restrict__ res) {
  long i = (long)blockIdx.x * blockDim.x + threadIdx.x;
  if (i >= total) return;
  int c = (int)((i / S) % C);
  float v = (x[i] - mean[c]) * rsqrtf(var[c] + 1e-5f) * g[c] + beta[c];
  if (act == 1) v = (v >= 0.f) ? v : 0.2f * v;
  else if (act == 2) v = fmaxf(v, 0.f);
  if (res) v += res[i];
  y[i] = v;
}

// ---------------------------------------------------------------------------
// MaxPool3d with window = stride
// ---------------------------------------------------------------------------
__global__ void k_maxpool3d(const float* __restrict__ x, float* __restrict__ y,
                            long NBC, int D, int H, int W, int wd, int wh, int ww) {
  int Do = D / wd, Ho = H / wh, Wo = W / ww;
  long total = NBC * (long)Do * Ho * Wo;
  long idx = (long)blockIdx.x * blockDim.x + threadIdx.x;
  if (idx >= total) return;
  long t = idx;
  int ow = (int)(t % Wo); t /= Wo;
  int oh = (int)(t % Ho); t /= Ho;
  int od = (int)(t % Do); long nc = t / Do;
  const float* xb = x + nc * (long)D * H * W;
  float m = -INFINITY;
  for (int a = 0; a < wd; ++a)
    for (int b = 0; b < wh; ++b)
      for (int c = 0; c < ww; ++c)
        m = fmaxf(m, xb[((long)(od * wd + a) * H + (oh * wh + b)) * W + (ow * ww + c)]);
  y[idx] = m;
}

// ---------------------------------------------------------------------------
// ConvTranspose3d (gather form).  w: [Cin,Cout,kd,kh,kw]
// ---------------------------------------------------------------------------
__global__ void k_deconv3d(const float* __restrict__ x, const float* __restrict__ w,
                           float* __restrict__ y, int NB, int Cin, int Cout,
                           int Di, int Hi, int Wi, int Do, int Ho, int Wo,
                           int kd, int kh, int kw, int sd, int sh, int sw,
                           int pd, int ph, int pw) {
  long total = (long)NB * Cout * Do * Ho * Wo;
  long idx = (long)blockIdx.x * blockDim.x + threadIdx.x;
  if (idx >= total) return;
  long t = idx;
  int ow = (int)(t % Wo); t /= Wo;
  int oh = (int)(t % Ho); t /= Ho;
  int od = (int)(t % Do); t /= Do;
  int co = (int)(t % Cout); int nb = (int)(t / Cout);
  float acc = 0.f;
  for (int ci = 0; ci < Cin; ++ci) {
    const float* xb = x + ((long)nb * Cin + ci) * Di * Hi * Wi;
    const float* wb = w + ((long)ci * Cout + co) * kd * kh * kw;
    for (int a = 0; a < kd; ++a) {
      int td = od + pd - a; if (td < 0 || td % sd) continue;
      int id = td / sd; if (id >= Di) continue;
      for (int b = 0; b < kh; ++b) {
        int th = oh + ph - b; if (th < 0 || th % sh) continue;
        int ih = th / sh; if (ih >= Hi) continue;
        for (int c = 0; c < kw; ++c) {
          int tw = ow + pw - c; if (tw < 0 || tw % sw) continue;
          int iw = tw / sw; if (iw >= Wi) continue;
          acc += xb[((long)id * Hi + ih) * Wi + iw] * wb[(a * kh + b) * kw + c];
        }
      }
    }
  }
  y[idx] = acc;
}

// ---------------------------------------------------------------------------
// Gridding-reverse: per voxel, density-weighted mean of its 8 corner coords
// ---------------------------------------------------------------------------
__global__ void k_gridding_reverse(const float* __restrict__ grid,
                                   float* __restrict__ pts) {
  long idx = (long)blockIdx.x * blockDim.x + threadIdx.x;
  if (idx >= 4L * 128 * 128 * 16) return;
  long t = idx;
  int z = (int)(t % 16);  t /= 16;
  int y = (int)(t % 128); t /= 128;
  int x = (int)(t % 128); int b = (int)(t / 128);
  float wsum = 0.f, nx = 0.f, ny = 0.f, nz = 0.f;
  for (int c = 0; c < 8; ++c) {
    int i = c >> 2, j = (c >> 1) & 1, k = c & 1;
    int xx = x + i, yy = y + j, zz = z + k;
    float v = (xx < 128 && yy < 128 && zz < 16)
                  ? grid[(((long)b * 128 + xx) * 128 + yy) * 16 + zz] : 0.f;
    wsum += v;
    nx += v * ((float)xx / 128.f * 2.f - 1.f);
    ny += v * ((float)yy / 128.f * 2.f - 1.f);
    nz += v * ((float)zz / 16.f * 2.f - 1.f);
  }
  float* o = pts + idx * 3;
  if (wsum > 1e-8f) {
    float inv = 1.f / fmaxf(wsum, 1e-8f);
    o[0] = nx * inv; o[1] = ny * inv; o[2] = nz * inv;
  } else { o[0] = 0.f; o[1] = 0.f; o[2] = 0.f; }
}

// Deterministic analog of RandomPointSampling: per output slot, first nonzero
// point inside its stride-128 block of the 262144-point cloud.
__global__ void k_sample_points(const float* __restrict__ sparse,
                                float* __restrict__ sampled) {
  int idx = blockIdx.x * blockDim.x + threadIdx.x;   // 4*2048
  if (idx >= 4 * 2048) return;
  int b = idx >> 11, i = idx & 2047;
  const float* base = sparse + ((long)b * 262144 + (long)i * 128) * 3;
  float px = 0.f, py = 0.f, pz = 0.f;
  for (int j = 0; j < 128; ++j) {
    float a = base[j * 3], bb = base[j * 3 + 1], c = base[j * 3 + 2];
    if (a != 0.f || bb != 0.f || c != 0.f) { px = a; py = bb; pz = c; break; }
  }
  sampled[(long)idx * 3]     = px;
  sampled[(long)idx * 3 + 1] = py;
  sampled[(long)idx * 3 + 2] = pz;
}

// Cubic feature sampling: gather features at the 8 cell corners of each point,
// writing into pf[b, n, colOff + corner*C + c] with row stride 1792.
__global__ void k_cubic_sample(const float* __restrict__ pts,
                               const float* __restrict__ feat,
                               float* __restrict__ pf, int C,
                               int Dx, int Dy, int Dz, int colOff) {
  int idx = blockIdx.x * blockDim.x + threadIdx.x;   // 4*2048*8
  if (idx >= 4 * 2048 * 8) return;
  int corner = idx & 7;
  int pi = idx >> 3;
  int b = pi >> 11;
  const float* p = pts + (long)pi * 3;
  float gx = fminf(fmaxf((p[0] * 0.5f + 0.5f) * Dx, 0.f), (float)Dx - 1.000001f);
  float gy = fminf(fmaxf((p[1] * 0.5f + 0.5f) * Dy, 0.f), (float)Dy - 1.000001f);
  float gz = fminf(fmaxf((p[2] * 0.5f + 0.5f) * Dz, 0.f), (float)Dz - 1.000001f);
  int ix = (int)floorf(gx) + (corner >> 2);
  int iy = (int)floorf(gy) + ((corner >> 1) & 1);
  int iz = (int)floorf(gz) + (corner & 1);
  const float* fb = feat + (long)b * C * Dx * Dy * Dz;
  float* o = pf + (long)pi * 1792 + colOff + corner * C;
  for (int c = 0; c < C; ++c)
    o[c] = fb[(((long)c * Dx + ix) * Dy + iy) * Dz + iz];
}

// d_out = [sampled (4*2048*3) | dense = repeat(sampled,8)+offset (4*16384*3)]
__global__ void k_final_out(const float* __restrict__ sampled,
                            const float* __restrict__ offs,
                            float* __restrict__ out) {
  int idx = blockIdx.x * blockDim.x + threadIdx.x;
  int total = 24576 + 196608;
  if (idx >= total) return;
  if (idx < 24576) { out[idx] = sampled[idx]; return; }
  int i = idx - 24576;
  int c = i % 3; int q = i / 3;
  int n8 = q % 16384; int b = q / 16384; int n = n8 >> 3;
  out[idx] = sampled[((long)b * 2048 + n) * 3 + c] + offs[i];
}

// ---------------------------------------------------------------------------
// Host launcher
// ---------------------------------------------------------------------------
extern "C" void kernel_launch(void* const* d_in, const int* in_sizes, int n_in,
                              void* d_out, int out_size, void* d_ws, size_t ws_size,
                              hipStream_t stream) {
  (void)in_sizes; (void)n_in; (void)out_size; (void)ws_size;
  const float* pc    = (const float*)d_in[0];
  const float *w1 = (const float*)d_in[1],  *b1 = (const float*)d_in[2],
              *g1 = (const float*)d_in[3],  *be1 = (const float*)d_in[4];
  const float *w2 = (const float*)d_in[5],  *b2 = (const float*)d_in[6],
              *g2 = (const float*)d_in[7],  *be2 = (const float*)d_in[8];
  const float *w3 = (const float*)d_in[9],  *b3 = (const float*)d_in[10],
              *g3 = (const float*)d_in[11], *be3 = (const float*)d_in[12];
  const float *w4 = (const float*)d_in[13], *b4 = (const float*)d_in[14],
              *g4 = (const float*)d_in[15], *be4 = (const float*)d_in[16];
  const float *fc5w = (const float*)d_in[17], *fc5b = (const float*)d_in[18];
  const float *fc6w = (const float*)d_in[19], *fc6b = (const float*)d_in[20];
  const float *dw7 = (const float*)d_in[21], *g7 = (const float*)d_in[22],  *be7 = (const float*)d_in[23];
  const float *dw8 = (const float*)d_in[24], *g8 = (const float*)d_in[25],  *be8 = (const float*)d_in[26];
  const float *dw9 = (const float*)d_in[27], *g9 = (const float*)d_in[28],  *be9 = (const float*)d_in[29];
  const float *dw10 = (const float*)d_in[30], *g10 = (const float*)d_in[31], *be10 = (const float*)d_in[32];
  const float *fc11w = (const float*)d_in[33], *fc11b = (const float*)d_in[34];
  const float *fc12w = (const float*)d_in[35], *fc12b = (const float*)d_in[36];
  const float *fc13w = (const float*)d_in[37], *fc13b = (const float*)d_in[38];
  const float *fc14w = (const float*)d_in[39], *fc14b = (const float*)d_in[40];
  float* out = (float*)d_out;

  // bump allocator over d_ws
  size_t off = 0;
  auto alloc = [&](long nf) -> float* {
    float* p = (float*)((char*)d_ws + off);
    off += ((size_t)nf * sizeof(float) + 255) & ~(size_t)255;
    return p;
  };
  float* grid   = alloc(1048576);      // [4,1,128,128,16]
  float* c1     = alloc(33554432);     // [4,32,128,128,16]
  float* f64b   = alloc(8388608);      // [4,32,64,64,16]
  float* c2     = alloc(16777216);     // [4,64,64,64,16]
  float* f32b   = alloc(2097152);      // [4,64,32,32,8]
  float* c3     = alloc(4194304);      // [4,128,32,32,8]
  float* f16b   = alloc(1048576);      // [4,128,16,16,8]
  float* c4     = alloc(2097152);      // [4,256,16,16,8]
  float* f8b    = alloc(262144);       // [4,256,8,8,4]
  float* feats  = alloc(8192);         // [4,2048]
  float* f8r    = alloc(262144);       // [4,256,8,8,4]
  float* d7     = alloc(1048576);      // [4,128,16,16,8]  -> f16r in place
  float* d8     = alloc(2097152);      // [4,64,32,32,8]   -> f32r in place
  float* d9     = alloc(8388608);      // [4,32,64,64,16]  -> f64r in place
  float* d10    = alloc(1048576);      // [4,1,128,128,16] -> xr in place
  float* sparse = alloc(3145728);      // [4,262144,3]
  float* smp    = alloc(24576);        // [4,2048,3]
  float* pf     = alloc(14680064);     // [8192,1792]
  float* pfb    = alloc(14680064);     // [8192,1792]
  float* offs   = alloc(196608);       // [8192,24]
  float* mean   = alloc(256);
  float* var    = alloc(256);

  auto blocks = [](long total) { return dim3((unsigned)((total + 255) / 256)); };
  auto conv = [&](const float* x, const float* w, const float* b, float* y,
                  int Cin, int Cout, int D, int H, int W) {
    long NN = 4L * D * H * W;
    long waves = ((Cout + 15) / 16) * ((NN + 15) / 16);
    k_conv3d_wmma<<<blocks(waves * 32), dim3(256), 0, stream>>>(x, w, b, y, 4, Cin, Cout, D, H, W);
  };
  auto gemm = [&](const float* A, const float* B, const float* bias, const float* res,
                  float* C, int M, int N, int K, int act) {
    long waves = (long)((M + 15) / 16) * ((N + 15) / 16);
    long threads = waves * 32;
    k_gemm_f16_wmma<<<dim3((unsigned)((threads + 127) / 128)), dim3(128), 0, stream>>>(
        A, B, bias, res, C, M, N, K, act);
  };
  auto bn = [&](float* x, const float* g, const float* be, int C, long S, int act,
                const float* res) {
    k_bn_stats<<<dim3(C), dim3(256), 0, stream>>>(x, mean, var, C, S, 4);
    long total = 4L * C * S;
    k_bn_act_res<<<blocks(total), dim3(256), 0, stream>>>(x, x, mean, var, g, be, C, S, total, act, res);
  };

  // ---- gridding ----
  k_fill_zero<<<blocks(1048576), dim3(256), 0, stream>>>(grid, 1048576);
  k_gridding<<<blocks(4 * 2048), dim3(256), 0, stream>>>(pc, grid);

  // ---- encoder ----
  conv(grid, w1, b1, c1, 1, 32, 128, 128, 16);
  bn(c1, g1, be1, 32, 128L * 128 * 16, 1, nullptr);
  k_maxpool3d<<<blocks(8388608), dim3(256), 0, stream>>>(c1, f64b, 4 * 32, 128, 128, 16, 2, 2, 1);

  conv(f64b, w2, b2, c2, 32, 64, 64, 64, 16);
  bn(c2, g2, be2, 64, 64L * 64 * 16, 1, nullptr);
  k_maxpool3d<<<blocks(2097152), dim3(256), 0, stream>>>(c2, f32b, 4 * 64, 64, 64, 16, 2, 2, 2);

  conv(f32b, w3, b3, c3, 64, 128, 32, 32, 8);
  bn(c3, g3, be3, 128, 32L * 32 * 8, 1, nullptr);
  k_maxpool3d<<<blocks(1048576), dim3(256), 0, stream>>>(c3, f16b, 4 * 128, 32, 32, 8, 2, 2, 1);

  conv(f16b, w4, b4, c4, 128, 256, 16, 16, 8);
  bn(c4, g4, be4, 256, 16L * 16 * 8, 1, nullptr);
  k_maxpool3d<<<blocks(262144), dim3(256), 0, stream>>>(c4, f8b, 4 * 256, 16, 16, 8, 2, 2, 2);

  // ---- bottleneck FCs (memory-bound: 536 MB weights each) ----
  gemm(f8b, fc5w, fc5b, nullptr, feats, 4, 2048, 65536, 2);          // relu
  gemm(feats, fc6w, fc6b, f8b, f8r, 4, 65536, 2048, 2);              // relu + skip

  // ---- decoder ----
  k_deconv3d<<<blocks(1048576), dim3(256), 0, stream>>>(f8r, dw7, d7, 4, 256, 128,
      8, 8, 4, 16, 16, 8, 4, 4, 4, 2, 2, 2, 1, 1, 1);
  bn(d7, g7, be7, 128, 16L * 16 * 8, 2, f16b);                       // -> f16r
  k_deconv3d<<<blocks(2097152), dim3(256), 0, stream>>>(d7, dw8, d8, 4, 128, 64,
      16, 16, 8, 32, 32, 8, 4, 4, 1, 2, 2, 1, 1, 1, 0);
  bn(d8, g8, be8, 64, 32L * 32 * 8, 2, f32b);                        // -> f32r
  k_deconv3d<<<blocks(8388608), dim3(256), 0, stream>>>(d8, dw9, d9, 4, 64, 32,
      32, 32, 8, 64, 64, 16, 4, 4, 4, 2, 2, 2, 1, 1, 1);
  bn(d9, g9, be9, 32, 64L * 64 * 16, 2, f64b);                       // -> f64r
  k_deconv3d<<<blocks(1048576), dim3(256), 0, stream>>>(d9, dw10, d10, 4, 32, 1,
      64, 64, 16, 128, 128, 16, 4, 4, 1, 2, 2, 1, 1, 1, 0);
  bn(d10, g10, be10, 1, 128L * 128 * 16, 2, grid);                   // -> xr

  // ---- points ----
  k_gridding_reverse<<<blocks(4L * 128 * 128 * 16), dim3(256), 0, stream>>>(d10, sparse);
  k_sample_points<<<blocks(4 * 2048), dim3(256), 0, stream>>>(sparse, smp);
  k_cubic_sample<<<blocks(4 * 2048 * 8), dim3(256), 0, stream>>>(smp, d9, pf, 32, 64, 64, 16, 0);
  k_cubic_sample<<<blocks(4 * 2048 * 8), dim3(256), 0, stream>>>(smp, d8, pf, 64, 32, 32, 8, 256);
  k_cubic_sample<<<blocks(4 * 2048 * 8), dim3(256), 0, stream>>>(smp, d7, pf, 128, 16, 16, 8, 768);

  // ---- point MLP ----
  gemm(pf,  fc11w, fc11b, nullptr, pfb,  8192, 1792, 1792, 2);
  gemm(pfb, fc12w, fc12b, nullptr, pf,   8192, 448,  1792, 2);
  gemm(pf,  fc13w, fc13b, nullptr, pfb,  8192, 112,  448,  2);
  gemm(pfb, fc14w, fc14b, nullptr, offs, 8192, 24,   112,  0);

  k_final_out<<<blocks(24576 + 196608), dim3(256), 0, stream>>>(smp, offs, out);
}
